// LightCrossAttention_67920612819605
// MI455X (gfx1250) — matrix operand order repaired and checked by reading, not compiled, not run on hardware
//
#include <hip/hip_runtime.h>
#include <hip/hip_bf16.h>

// ---------------------------------------------------------------------------
// LightCrossAttention for MI455X (gfx1250, wave32, WMMA)
// B=4, N=4096, DIM=360 (pad 384), RED=40 (pad 64)
// Big GEMMs: v_wmma_f32_16x16x32_bf16, double-buffered LDS staging,
// async memory->LDS (GLOBAL_LOAD_ASYNC_TO_LDS_B128, ASYNCcnt) where the
// operand needs no transpose; VGPR+ds_store scatter where it does.
// ---------------------------------------------------------------------------

typedef __attribute__((ext_vector_type(16))) __bf16 v16bf;
typedef __attribute__((ext_vector_type(8)))  __bf16 v8bf;
typedef __attribute__((ext_vector_type(4)))  __bf16 v4bf;
typedef __attribute__((ext_vector_type(2)))  __bf16 v2bf;
typedef __attribute__((ext_vector_type(8)))  float  v8f;

#define BATCH 4
#define SEQ   4096
#define DIM   360
#define DIMP  384
#define RED   40
#define REDP  64

__device__ __forceinline__ float gelu_exact(float x) {
    return 0.5f * x * (1.0f + erff(x * 0.70710678118654752f));
}

// async memory -> LDS, 16 bytes per lane, tracked on ASYNCcnt
__device__ __forceinline__ void async_copy_b128(unsigned lds_addr, const void* gaddr) {
    asm volatile("global_load_async_to_lds_b128 %0, %1, off"
                 :: "v"(lds_addr), "v"((unsigned long long)gaddr)
                 : "memory");
}
__device__ __forceinline__ void async_wait_all() {
    asm volatile("s_wait_asynccnt 0x0" ::: "memory");
}

// ---------------------------------------------------------------------------
// 1) float -> bf16 with column pad 360 -> 384 (zeros)
// ---------------------------------------------------------------------------
__global__ __launch_bounds__(256)
void cvt_pad_bf16_kernel(const float* __restrict__ X, __bf16* __restrict__ Y,
                         long long rows) {
    long long idx = (long long)blockIdx.x * blockDim.x + threadIdx.x;
    long long total = rows * DIMP;
    if (idx >= total) return;
    int col = (int)(idx % DIMP);
    long long row = idx / DIMP;
    float v = (col < DIM) ? X[row * DIM + col] : 0.0f;
    Y[idx] = (__bf16)v;
}

// ---------------------------------------------------------------------------
// 2) reduction MLP: Y[row,k] = gelu(X[row,:] @ W[:,k] + b[k]), K padded to 64
// ---------------------------------------------------------------------------
__global__ __launch_bounds__(64)
void reduce_gelu_kernel(const float* __restrict__ X, const float* __restrict__ W,
                        const float* __restrict__ bias, __bf16* __restrict__ Y) {
    const long long row = blockIdx.x;
    __shared__ float xrow[DIM];
    for (int i = threadIdx.x; i < DIM; i += 64) xrow[i] = X[row * DIM + i];
    __syncthreads();
    const int k = threadIdx.x;
    __bf16 out = (__bf16)0.0f;
    if (k < RED) {
        float acc = bias[k];
        for (int d = 0; d < DIM; ++d) acc = fmaf(xrow[d], W[d * RED + k], acc);
        out = (__bf16)gelu_exact(acc);
    }
    Y[row * REDP + k] = out;   // pad K 40..63 with zeros
}

// ---------------------------------------------------------------------------
// 3) Tiled bf16 WMMA GEMM:  C[M x Nn] (f32) = opA(A) * opB(B)
//    TRANS_A:    memory holds A[k][m] (lda strides k)  -> sync LDS transpose
//    B_COLMAJOR: memory holds Bm[n][k] (ldb strides n) -> async direct copy
//    !B_COLMAJOR: memory holds Bm[k][n] (ldb strides k)-> sync LDS transpose
//    Block tile 128x128, BK=32, 256 threads = 8 waves, wave tile 64x32.
//    Double-buffered LDS; async issues overlap with WMMA of current buffer.
// ---------------------------------------------------------------------------
template<bool TRANS_A, bool B_COLMAJOR>
__global__ __launch_bounds__(256)
void gemm_bf16_wmma(const __bf16* __restrict__ A, const __bf16* __restrict__ Bm,
                    float* __restrict__ C,
                    int M, int Nn, int K, int lda, int ldb, int ldc,
                    long long strideA, long long strideB, long long strideC) {
    __shared__ __align__(16) __bf16 As[2][128][40];   // [buf][m][k], padded row
    __shared__ __align__(16) __bf16 Bs[2][128][40];   // [buf][n][k], padded row

    const int tid  = threadIdx.x;
    const int lane = tid & 31;
    const int wave = tid >> 5;
    const int wm   = wave & 1;   // 2 waves along M
    const int wn   = wave >> 1;  // 4 waves along N
    const int m_blk = blockIdx.y * 128;
    const int n_blk = blockIdx.x * 128;
    const int h   = lane >> 4;
    const int l16 = lane & 15;

    A  += (long long)blockIdx.z * strideA;
    Bm += (long long)blockIdx.z * strideB;
    C  += (long long)blockIdx.z * strideC;

    // stage one BK=32 slab of A and B into LDS buffer `buf`
    auto stage = [&](int buf, int kb) {
        if (!TRANS_A) {
            const int row = tid >> 1, seg = tid & 1;
            const __bf16* src = A + (long long)(m_blk + row) * lda + kb + seg * 16;
            unsigned laddr = (unsigned)(size_t)(&As[buf][row][seg * 16]);
            async_copy_b128(laddr,      src);
            async_copy_b128(laddr + 16, src + 8);
        } else {
            const int krow = tid >> 3, mseg = tid & 7;
            const __bf16* src = A + (long long)(kb + krow) * lda + m_blk + mseg * 16;
            v8bf t0 = *(const v8bf*)(src);
            v8bf t1 = *(const v8bf*)(src + 8);
#pragma unroll
            for (int i = 0; i < 8; ++i) As[buf][mseg * 16 + i][krow]     = t0[i];
#pragma unroll
            for (int i = 0; i < 8; ++i) As[buf][mseg * 16 + 8 + i][krow] = t1[i];
        }
        if (B_COLMAJOR) {
            const int nrow = tid >> 1, seg = tid & 1;
            const __bf16* src = Bm + (long long)(n_blk + nrow) * ldb + kb + seg * 16;
            unsigned laddr = (unsigned)(size_t)(&Bs[buf][nrow][seg * 16]);
            async_copy_b128(laddr,      src);
            async_copy_b128(laddr + 16, src + 8);
        } else {
            const int krow = tid >> 3, nseg = tid & 7;
            const __bf16* src = Bm + (long long)(kb + krow) * ldb + n_blk + nseg * 16;
            v8bf t0 = *(const v8bf*)(src);
            v8bf t1 = *(const v8bf*)(src + 8);
#pragma unroll
            for (int i = 0; i < 8; ++i) Bs[buf][nseg * 16 + i][krow]     = t0[i];
#pragma unroll
            for (int i = 0; i < 8; ++i) Bs[buf][nseg * 16 + 8 + i][krow] = t1[i];
        }
    };

    const v8f vzero = {};
    v8f acc[4][2];
#pragma unroll
    for (int ai = 0; ai < 4; ++ai)
#pragma unroll
        for (int bj = 0; bj < 2; ++bj) acc[ai][bj] = vzero;

    const int nsteps = K >> 5;          // K multiple of 32
    stage(0, 0);                        // prologue

    for (int i = 0; i < nsteps; ++i) {
        const int cur = i & 1;
        async_wait_all();               // my async writes into LDS[cur] landed
        __syncthreads();                // everyone's writes landed; prev reads done
        if (i + 1 < nsteps) stage(cur ^ 1, (i + 1) * 32);   // overlap with WMMA

        // ---- build B fragments (lane n=l16, VGPR v holds K=(v/4)*16 + h*8 + (v%4)*2,+1)
        v16bf bfrag[2];
#pragma unroll
        for (int bj = 0; bj < 2; ++bj) {
            const int n0 = wn * 32 + bj * 16 + l16;
#pragma unroll
            for (int v = 0; v < 8; ++v) {
                const int K0 = (v >> 2) * 16 + h * 8 + (v & 3) * 2;
                v2bf t = *(const v2bf*)(&Bs[cur][n0][K0]);
                bfrag[bj][2 * v]     = t[0];
                bfrag[bj][2 * v + 1] = t[1];
            }
        }
        // ---- A fragments + WMMA (A lane m=l16; VGPR v: K=(v/4)*16 + h*8 + (v%4)*2)
#pragma unroll
        for (int ai = 0; ai < 4; ++ai) {
            const int m0 = wm * 64 + ai * 16 + l16;
            v16bf afrag;
#pragma unroll
            for (int v = 0; v < 8; ++v) {
                const int K0 = (v >> 2) * 16 + h * 8 + (v & 3) * 2;
                v2bf t = *(const v2bf*)(&As[cur][m0][K0]);
                afrag[2 * v]     = t[0];
                afrag[2 * v + 1] = t[1];
            }
#pragma unroll
            for (int bj = 0; bj < 2; ++bj) {
                acc[ai][bj] = __builtin_amdgcn_wmma_f32_16x16x32_bf16(
                    false, afrag, false, bfrag[bj], (short)0, acc[ai][bj],
                    false, false);
            }
        }
    }

    // ---- epilogue: D layout — VGPR r holds M = r + 8*h, N = l16 ----
#pragma unroll
    for (int ai = 0; ai < 4; ++ai) {
#pragma unroll
        for (int bj = 0; bj < 2; ++bj) {
            const int n = n_blk + wn * 32 + bj * 16 + l16;
            if (n < Nn) {
                const int mbase = m_blk + wm * 64 + ai * 16 + h * 8;
#pragma unroll
                for (int r = 0; r < 8; ++r) {
                    C[(long long)(mbase + r) * ldc + n] = acc[ai][bj][r];
                }
            }
        }
    }
}

// ---------------------------------------------------------------------------
// 4) row softmax: f32 logits -> normalized bf16 attention
// ---------------------------------------------------------------------------
__global__ __launch_bounds__(256)
void softmax_bf16_kernel(const float* __restrict__ logits,
                         __bf16* __restrict__ attn) {
    const long long row = (long long)blockIdx.y * gridDim.x + blockIdx.x;
    const float* x = logits + row * SEQ;
    __bf16* y = attn + row * SEQ;
    __shared__ float red[256];
    const int tid = threadIdx.x;

    float4 vals[4];
    float m = -3.0e38f;
#pragma unroll
    for (int j = 0; j < 4; ++j) {
        vals[j] = ((const float4*)x)[tid + 256 * j];
        m = fmaxf(m, fmaxf(fmaxf(vals[j].x, vals[j].y), fmaxf(vals[j].z, vals[j].w)));
    }
    red[tid] = m;
    __syncthreads();
    for (int s = 128; s > 0; s >>= 1) {
        if (tid < s) red[tid] = fmaxf(red[tid], red[tid + s]);
        __syncthreads();
    }
    m = red[0];
    __syncthreads();

    float sum = 0.0f;
#pragma unroll
    for (int j = 0; j < 4; ++j) {
        vals[j].x = __expf(vals[j].x - m);
        vals[j].y = __expf(vals[j].y - m);
        vals[j].z = __expf(vals[j].z - m);
        vals[j].w = __expf(vals[j].w - m);
        sum += vals[j].x + vals[j].y + vals[j].z + vals[j].w;
    }
    red[tid] = sum;
    __syncthreads();
    for (int s = 128; s > 0; s >>= 1) {
        if (tid < s) red[tid] += red[tid + s];
        __syncthreads();
    }
    const float inv = 1.0f / red[0];

#pragma unroll
    for (int j = 0; j < 4; ++j) {
        v4bf o;
        o[0] = (__bf16)(vals[j].x * inv);
        o[1] = (__bf16)(vals[j].y * inv);
        o[2] = (__bf16)(vals[j].z * inv);
        o[3] = (__bf16)(vals[j].w * inv);
        *(v4bf*)(&y[(tid + 256 * j) * 4]) = o;
    }
}

// ---------------------------------------------------------------------------
// 5) output MLPs (GELU) on both branches, summed
// ---------------------------------------------------------------------------
__global__ __launch_bounds__(128)
void mlp_out_kernel(const float* __restrict__ fs_all, const float* __restrict__ ff_all,
                    const float* __restrict__ sp_w1, const float* __restrict__ sp_b1,
                    const float* __restrict__ sp_w2, const float* __restrict__ sp_b2,
                    const float* __restrict__ fr_w1, const float* __restrict__ fr_b1,
                    const float* __restrict__ fr_w2, const float* __restrict__ fr_b2,
                    float* __restrict__ out) {
    const long long row = blockIdx.x;
    __shared__ float fs[DIM], ff[DIM], ha[RED], hb[RED];
    for (int i = threadIdx.x; i < DIM; i += 128) {
        fs[i] = fs_all[row * DIM + i];
        ff[i] = ff_all[row * DIM + i];
    }
    __syncthreads();
    const int t = threadIdx.x;
    if (t < RED) {
        float acc = sp_b1[t];
        for (int d = 0; d < DIM; ++d) acc = fmaf(fs[d], sp_w1[d * RED + t], acc);
        ha[t] = gelu_exact(acc);
    } else if (t >= 64 && t < 64 + RED) {
        const int k = t - 64;
        float acc = fr_b1[k];
        for (int d = 0; d < DIM; ++d) acc = fmaf(ff[d], fr_w1[d * RED + k], acc);
        hb[k] = gelu_exact(acc);
    }
    __syncthreads();
    for (int d = t; d < DIM; d += 128) {
        float acc = sp_b2[d] + fr_b2[d];
        for (int k = 0; k < RED; ++k)
            acc = fmaf(ha[k], sp_w2[k * DIM + d], fmaf(hb[k], fr_w2[k * DIM + d], acc));
        out[row * DIM + d] = acc;
    }
}

// ---------------------------------------------------------------------------
extern "C" void kernel_launch(void* const* d_in, const int* in_sizes, int n_in,
                              void* d_out, int out_size, void* d_ws, size_t ws_size,
                              hipStream_t stream) {
    (void)in_sizes; (void)n_in; (void)out_size; (void)ws_size;
    const float* spatial = (const float*)d_in[0];
    const float* freq    = (const float*)d_in[1];
    const float* red_w   = (const float*)d_in[2];
    const float* red_b   = (const float*)d_in[3];
    const float* sp_w1   = (const float*)d_in[4];
    const float* sp_b1   = (const float*)d_in[5];
    const float* sp_w2   = (const float*)d_in[6];
    const float* sp_b2   = (const float*)d_in[7];
    const float* fr_w1   = (const float*)d_in[8];
    const float* fr_b1   = (const float*)d_in[9];
    const float* fr_w2   = (const float*)d_in[10];
    const float* fr_b2   = (const float*)d_in[11];
    float* out = (float*)d_out;

    const long long rows = (long long)BATCH * SEQ;   // 16384

    // workspace carve-up (256B aligned)
    char* ws = (char*)d_ws;
    size_t off = 0;
    auto carve = [&](size_t bytes) {
        void* p = ws + off;
        off += (bytes + 255) & ~(size_t)255;
        return p;
    };
    float*  logits = (float*) carve((size_t)BATCH * SEQ * SEQ * 4);   // 268 MB
    __bf16* attn   = (__bf16*)carve((size_t)BATCH * SEQ * SEQ * 2);   // 134 MB (L2-resident)
    __bf16* Sr     = (__bf16*)carve((size_t)rows * REDP * 2);
    __bf16* Fr     = (__bf16*)carve((size_t)rows * REDP * 2);
    __bf16* sp_bf  = (__bf16*)carve((size_t)rows * DIMP * 2);
    __bf16* fr_bf  = (__bf16*)carve((size_t)rows * DIMP * 2);
    float*  fus_sp = (float*) carve((size_t)rows * DIM * 4);
    float*  fus_fr = (float*) carve((size_t)rows * DIM * 4);

    // 1) pad/convert spatial & freq to bf16 [rows x 384]
    {
        long long total = rows * DIMP;
        int nb = (int)((total + 255) / 256);
        cvt_pad_bf16_kernel<<<nb, 256, 0, stream>>>(spatial, sp_bf, rows);
        cvt_pad_bf16_kernel<<<nb, 256, 0, stream>>>(freq,    fr_bf, rows);
    }
    // 2) reduction MLP + GELU -> Sr, Fr (bf16, K padded to 64)
    reduce_gelu_kernel<<<(int)rows, 64, 0, stream>>>(spatial, red_w, red_b, Sr);
    reduce_gelu_kernel<<<(int)rows, 64, 0, stream>>>(freq,    red_w, red_b, Fr);

    // 3) attn logits: Sr [4096x64] x Fr^T -> f32 [4096x4096] per batch
    gemm_bf16_wmma<false, true><<<dim3(SEQ / 128, SEQ / 128, BATCH), 256, 0, stream>>>(
        Sr, Fr, logits, SEQ, SEQ, REDP, REDP, REDP, SEQ,
        (long long)SEQ * REDP, (long long)SEQ * REDP, (long long)SEQ * SEQ);

    // 4) row softmax -> bf16 attention weights
    softmax_bf16_kernel<<<dim3(SEQ, BATCH), 256, 0, stream>>>(logits, attn);

    // 5) fused_spatial = attn @ freq      (A row-major async, B k-major transpose)
    gemm_bf16_wmma<false, false><<<dim3(DIMP / 128, SEQ / 128, BATCH), 256, 0, stream>>>(
        attn, fr_bf, fus_sp, SEQ, DIM, SEQ, SEQ, DIMP, DIM,
        (long long)SEQ * SEQ, (long long)SEQ * DIMP, (long long)SEQ * DIM);

    // 6) fused_freq = attn^T @ spatial    (A transposed via LDS staging)
    gemm_bf16_wmma<true, false><<<dim3(DIMP / 128, SEQ / 128, BATCH), 256, 0, stream>>>(
        attn, sp_bf, fus_fr, SEQ, DIM, SEQ, SEQ, DIMP, DIM,
        (long long)SEQ * SEQ, (long long)SEQ * DIMP, (long long)SEQ * DIM);

    // 7) output MLPs + sum -> out (f32)
    mlp_out_kernel<<<(int)rows, 128, 0, stream>>>(
        fus_sp, fus_fr, sp_w1, sp_b1, sp_w2, sp_b2,
        fr_w1, fr_b1, fr_w2, fr_b2, out);
}
